// HybridClassifier_65481071402826
// MI455X (gfx1250) — matrix operand order repaired and checked by reading, not compiled, not run on hardware
//
#include <hip/hip_runtime.h>

// ---------------------------------------------------------------------------
// Shapes
// ---------------------------------------------------------------------------
#define BATCH   128
// conv1: in (128,3,250,250) -> conv (128,6,124,124) -> pool (128,6,123,123)
#define C1_OUT_H 123
// conv2: in (128,6,123,123) -> conv (128,15,62,62) -> pool (128,15,61,61)
#define C2_OUT_H 61
#define FC1_K    55815          // 15*61*61
#define FC1_KP   57344          // padded: 32 * 1792; 1792 = 16 splits * 28 iters * 4 unroll
#define FC1_KIT  (FC1_KP / 32)  // 1792 wmma k-steps
#define FC1_N    120
#define KSPLIT   16
#define KCHUNK   112            // 1792 / 16, multiple of UNROLL
#define UNROLL   4

typedef __attribute__((ext_vector_type(16))) __bf16        v16bf;
typedef __attribute__((ext_vector_type(8)))  float         v8f;
typedef __attribute__((ext_vector_type(4)))  unsigned int  uint4v;

__device__ __forceinline__ unsigned short f32_to_bf16(float f) {
    unsigned int u = __float_as_uint(f);
    u += 0x7FFFu + ((u >> 16) & 1u);   // round to nearest even
    return (unsigned short)(u >> 16);
}
__device__ __forceinline__ float bf16_to_f32(unsigned short h) {
    return __uint_as_float(((unsigned int)h) << 16);
}

// ---------------------------------------------------------------------------
// Kernel 0: zero the padded tail columns of the activation matrix
// ---------------------------------------------------------------------------
#define TAILN (FC1_KP - FC1_K)   // 1529
__global__ void init_tail(unsigned short* __restrict__ actA) {
    int i = blockIdx.x * 256 + threadIdx.x;
    if (i < BATCH * TAILN) {
        int b = i / TAILN, j = i % TAILN;
        actA[(size_t)b * FC1_KP + FC1_K + j] = 0;
    }
}

// ---------------------------------------------------------------------------
// Kernel 1: conv1 (3->6, 5x5, s2, p1) + ReLU + 2x2/s1 maxpool, bf16 out
// ---------------------------------------------------------------------------
__global__ __launch_bounds__(256) void conv1_pool(
    const float* __restrict__ in, const float* __restrict__ wt,
    const float* __restrict__ bias, unsigned short* __restrict__ out) {
    int oc = blockIdx.y, b = blockIdx.z;
    int p = blockIdx.x * 256 + threadIdx.x;
    if (p >= C1_OUT_H * C1_OUT_H) return;
    int ph = p / C1_OUT_H, pw = p % C1_OUT_H;
    const float* W = wt + oc * 75;                 // uniform -> scalar loads
    const float* I = in + (size_t)b * 3 * 250 * 250;
    float mx = -1e30f;
    #pragma unroll
    for (int dh = 0; dh < 2; ++dh)
    #pragma unroll
    for (int dw = 0; dw < 2; ++dw) {
        int h = ph + dh, w = pw + dw;
        float acc = bias[oc];
        for (int ic = 0; ic < 3; ++ic)
        #pragma unroll
        for (int kh = 0; kh < 5; ++kh) {
            int iy = 2 * h - 1 + kh;
            if (iy < 0 || iy >= 250) continue;
            const float* row = I + ((size_t)ic * 250 + iy) * 250;
            #pragma unroll
            for (int kw = 0; kw < 5; ++kw) {
                int ix = 2 * w - 1 + kw;
                if (ix < 0 || ix >= 250) continue;
                acc += row[ix] * W[(ic * 5 + kh) * 5 + kw];
            }
        }
        mx = fmaxf(mx, acc);
    }
    out[(((size_t)b * 6 + oc) * C1_OUT_H + ph) * C1_OUT_H + pw] =
        f32_to_bf16(fmaxf(mx, 0.f));
}

// ---------------------------------------------------------------------------
// Kernel 2: conv2 (6->15, 3x3, s2, p1) + ReLU + maxpool; writes the flattened
// fc1 activation matrix (bf16, row stride FC1_KP) directly -> free im2col
// ---------------------------------------------------------------------------
__global__ __launch_bounds__(256) void conv2_pool(
    const unsigned short* __restrict__ in, const float* __restrict__ wt,
    const float* __restrict__ bias, unsigned short* __restrict__ actA) {
    int oc = blockIdx.y, b = blockIdx.z;
    int p = blockIdx.x * 256 + threadIdx.x;
    if (p >= C2_OUT_H * C2_OUT_H) return;
    int ph = p / C2_OUT_H, pw = p % C2_OUT_H;
    const float* W = wt + oc * 54;
    const unsigned short* I = in + (size_t)b * 6 * C1_OUT_H * C1_OUT_H;
    float mx = -1e30f;
    #pragma unroll
    for (int dh = 0; dh < 2; ++dh)
    #pragma unroll
    for (int dw = 0; dw < 2; ++dw) {
        int h = ph + dh, w = pw + dw;
        float acc = bias[oc];
        for (int ic = 0; ic < 6; ++ic)
        #pragma unroll
        for (int kh = 0; kh < 3; ++kh) {
            int iy = 2 * h - 1 + kh;
            if (iy < 0 || iy >= C1_OUT_H) continue;
            const unsigned short* row = I + ((size_t)ic * C1_OUT_H + iy) * C1_OUT_H;
            #pragma unroll
            for (int kw = 0; kw < 3; ++kw) {
                int ix = 2 * w - 1 + kw;
                if (ix < 0 || ix >= C1_OUT_H) continue;
                acc += bf16_to_f32(row[ix]) * W[(ic * 3 + kh) * 3 + kw];
            }
        }
        mx = fmaxf(mx, acc);
    }
    actA[(size_t)b * FC1_KP + oc * 3721 + ph * C2_OUT_H + pw] =
        f32_to_bf16(fmaxf(mx, 0.f));
}

// ---------------------------------------------------------------------------
// Kernel 3: fc1_w (120 x 55815 f32) -> bf16, zero-padded to 128 x 57344
// ---------------------------------------------------------------------------
__global__ void convert_w(const float* __restrict__ W,
                          unsigned short* __restrict__ Wb) {
    size_t i = (size_t)blockIdx.x * 256 + threadIdx.x;
    if (i >= (size_t)128 * FC1_KP) return;
    int r = (int)(i / FC1_KP), k = (int)(i % FC1_KP);
    float v = (r < FC1_N && k < FC1_K) ? W[(size_t)r * FC1_K + k] : 0.f;
    Wb[i] = f32_to_bf16(v);
}

// ---------------------------------------------------------------------------
// Kernel 4: fc1 GEMM (128 x 57344) @ (57344 x 128^T), bf16 WMMA, f32 acc.
// 64 output tiles of 16x16, 16-way K-split => 1024 waves, each running a
// remainder-free 112-iteration K-loop, software-pipelined 4x: 16 B128 loads
// in flight per block while 4 WMMAs retire on partial loadcnt waits.
// Fragments per the CDNA5 16-bit A(16x32)/B(32x16) VGPR layouts.
// Partials are race-free per (ksplit,tile) slot -> deterministic.
// ---------------------------------------------------------------------------
__global__ __launch_bounds__(256) void fc1_wmma(
    const unsigned short* __restrict__ A,   // activations, 128 x FC1_KP
    const unsigned short* __restrict__ W,   // weights,     128 x FC1_KP
    float* __restrict__ P) {                // partials, KSPLIT x 128 x 120
    int wave = threadIdx.x >> 5;
    int lane = threadIdx.x & 31;
    int gw   = blockIdx.x * 8 + wave;       // 0..1023
    int tile = gw & 63;                     // 8 mt x 8 nt
    int ks   = gw >> 6;                     // 0..15 k-split slot
    int mt = tile >> 3, nt = tile & 7;

    int kbeg = ks * KCHUNK;
    int kend = kbeg + KCHUNK;               // exact: 16*112 == 1792 == FC1_KIT

    int m      = lane & 15;                 // A: M index / B: N index
    int khalfA = (lane >> 4) * 8;           // A: lanes16-31 hold K+8 / K+24
    int kbB    = (lane >> 4) * 8;           // B: lanes16-31 hold K+8 / K+24
    const unsigned short* arow = A + (size_t)(mt * 16 + m) * FC1_KP;
    const unsigned short* wrow = W + (size_t)(nt * 16 + m) * FC1_KP;

    v8f c = {};
    for (int ki = kbeg; ki < kend; ki += UNROLL) {
        union { uint4v u[2]; v16bf v; } a[UNROLL], b[UNROLL];
        #pragma unroll
        for (int u = 0; u < UNROLL; ++u) {
            int k = (ki + u) * 32;
            a[u].u[0] = *(const uint4v*)(arow + k + khalfA);       // K khalf..+7
            a[u].u[1] = *(const uint4v*)(arow + k + khalfA + 16);  // K 16+khalf..+7
            b[u].u[0] = *(const uint4v*)(wrow + k + kbB);          // K kbB..+7
            b[u].u[1] = *(const uint4v*)(wrow + k + kbB + 16);     // K 16+kbB..+7
        }
        __builtin_prefetch((const void*)(arow + (ki + UNROLL) * 32), 0, 1);
        __builtin_prefetch((const void*)(wrow + (ki + UNROLL) * 32), 0, 1);
        #pragma unroll
        for (int u = 0; u < UNROLL; ++u)
            c = __builtin_amdgcn_wmma_f32_16x16x32_bf16(
                    false, a[u].v, false, b[u].v, (short)0, c, false, false);
    }

    // D layout: VGPR r -> lanes0-15: M=r, N=lane; lanes16-31: M=r+8, N=lane-16
    int col = nt * 16 + (lane & 15);
    if (col < FC1_N) {
        float* Pks = P + (size_t)ks * BATCH * FC1_N;
        int rbase = mt * 16 + ((lane >> 4) << 3);
        #pragma unroll
        for (int r = 0; r < 8; ++r)
            Pks[(size_t)(rbase + r) * FC1_N + col] = c[r];
    }
}

// ---------------------------------------------------------------------------
// Kernel 5: reduce the 16 K-split partials
// ---------------------------------------------------------------------------
__global__ void fc1_reduce(const float* __restrict__ P, float* __restrict__ C) {
    int i = blockIdx.x * 256 + threadIdx.x;
    if (i >= BATCH * FC1_N) return;
    float acc = 0.f;
    #pragma unroll
    for (int s = 0; s < KSPLIT; ++s) acc += P[(size_t)s * BATCH * FC1_N + i];
    C[i] = acc;
}

// ---------------------------------------------------------------------------
// Quantum head: 4 qubits, RY(t) init, 2 layers of Rot + CNOT chain, <Z0>
// ---------------------------------------------------------------------------
__device__ void quantum_head(float t, const float* __restrict__ qp,
                             float* p0, float* p1) {
    float re[16], im[16];
    float cc = cosf(0.5f * t), ss = sinf(0.5f * t);
    #pragma unroll
    for (int i = 0; i < 16; ++i) {
        float a = 1.f;
        #pragma unroll
        for (int q = 0; q < 4; ++q) a *= ((i >> (3 - q)) & 1) ? ss : cc;
        re[i] = a; im[i] = 0.f;
    }
    for (int layer = 0; layer < 2; ++layer) {
        for (int q = 0; q < 4; ++q) {
            const float* g = qp + (layer * 4 + q) * 3;
            float phi = g[0], th = g[1], om = g[2];
            float ch = cosf(0.5f * th), sh = sinf(0.5f * th);
            float hs = 0.5f * (phi + om), hd = 0.5f * (phi - om);
            float m00r =  cosf(hs) * ch, m00i = -sinf(hs) * ch;
            float m01r = -cosf(hd) * sh, m01i = -sinf(hd) * sh;
            float m10r =  cosf(hd) * sh, m10i = -sinf(hd) * sh;
            float m11r =  cosf(hs) * ch, m11i =  sinf(hs) * ch;
            int bit = 1 << (3 - q);
            #pragma unroll
            for (int i = 0; i < 16; ++i) {
                if (i & bit) continue;
                int j = i | bit;
                float ar = re[i], ai = im[i], br = re[j], bi = im[j];
                re[i] = m00r * ar - m00i * ai + m01r * br - m01i * bi;
                im[i] = m00r * ai + m00i * ar + m01r * bi + m01i * br;
                re[j] = m10r * ar - m10i * ai + m11r * br - m11i * bi;
                im[j] = m10r * ai + m10i * ar + m11r * bi + m11i * br;
            }
        }
        for (int q = 0; q < 3; ++q) {          // CNOT(q, q+1)
            int cb = 1 << (3 - q), tb = 1 << (2 - q);
            #pragma unroll
            for (int i = 0; i < 16; ++i) {
                if ((i & cb) && !(i & tb)) {
                    int j = i | tb;
                    float tr = re[i]; re[i] = re[j]; re[j] = tr;
                    float ti = im[i]; im[i] = im[j]; im[j] = ti;
                }
            }
        }
    }
    float z = 0.f;
    #pragma unroll
    for (int i = 0; i < 16; ++i) {
        float p = re[i] * re[i] + im[i] * im[i];
        z += (i & 8) ? -p : p;                 // qubit0 = msb
    }
    *p0 = 0.5f * (z + 1.f);
    *p1 = 1.f - *p0;
}

// ---------------------------------------------------------------------------
// Kernel 6: fc2 + fc3 + quantum head, one block per batch row
// ---------------------------------------------------------------------------
__global__ __launch_bounds__(128) void head(
    const float* __restrict__ fc1o, const float* __restrict__ fc1_b,
    const float* __restrict__ fc2_w, const float* __restrict__ fc2_b,
    const float* __restrict__ fc3_w, const float* __restrict__ fc3_b,
    const float* __restrict__ qparams, float* __restrict__ out) {
    __shared__ float xs[FC1_N];
    __shared__ float ys[84];
    int b = blockIdx.x, tid = threadIdx.x;
    if (tid < FC1_N)
        xs[tid] = fmaxf(fc1o[b * FC1_N + tid] + fc1_b[tid], 0.f);
    __syncthreads();
    if (tid < 84) {
        float acc = fc2_b[tid];
        for (int k = 0; k < FC1_N; ++k) acc += fc2_w[tid * FC1_N + k] * xs[k];
        ys[tid] = fmaxf(acc, 0.f);
    }
    __syncthreads();
    if (tid == 0) {
        float t = fc3_b[0];
        for (int k = 0; k < 84; ++k) t += fc3_w[k] * ys[k];
        float p0, p1;
        quantum_head(t, qparams, &p0, &p1);
        out[b * 2 + 0] = p0;
        out[b * 2 + 1] = p1;
    }
}

// ---------------------------------------------------------------------------
// Host launcher
// ---------------------------------------------------------------------------
extern "C" void kernel_launch(void* const* d_in, const int* in_sizes, int n_in,
                              void* d_out, int out_size, void* d_ws, size_t ws_size,
                              hipStream_t stream) {
    const float* inputs  = (const float*)d_in[0];
    const float* conv1_w = (const float*)d_in[1];
    const float* conv1_b = (const float*)d_in[2];
    const float* conv2_w = (const float*)d_in[3];
    const float* conv2_b = (const float*)d_in[4];
    const float* fc1_w   = (const float*)d_in[5];
    const float* fc1_b   = (const float*)d_in[6];
    const float* fc2_w   = (const float*)d_in[7];
    const float* fc2_b   = (const float*)d_in[8];
    const float* fc3_w   = (const float*)d_in[9];
    const float* fc3_b   = (const float*)d_in[10];
    const float* qparams = (const float*)d_in[11];
    float* out = (float*)d_out;

    char* ws = (char*)d_ws;
    size_t off = 0;
    auto carve = [&](size_t bytes) {
        size_t o = off;
        off = (off + bytes + 255) & ~(size_t)255;
        return o;
    };
    unsigned short* buf1 = (unsigned short*)(ws + carve((size_t)BATCH * 6 * C1_OUT_H * C1_OUT_H * 2)); // 23.2 MB
    unsigned short* actA = (unsigned short*)(ws + carve((size_t)BATCH * FC1_KP * 2));                  // 14.7 MB
    unsigned short* wgtB = (unsigned short*)(ws + carve((size_t)128 * FC1_KP * 2));                    // 14.7 MB
    float*          bufP = (float*)(ws + carve((size_t)KSPLIT * BATCH * FC1_N * 4));                   // 0.98 MB
    float*          buf2 = (float*)(ws + carve((size_t)BATCH * FC1_N * 4));                            // 61 KB

    init_tail<<<(BATCH * TAILN + 255) / 256, 256, 0, stream>>>(actA);
    conv1_pool<<<dim3((C1_OUT_H * C1_OUT_H + 255) / 256, 6, BATCH), 256, 0, stream>>>(
        inputs, conv1_w, conv1_b, buf1);
    convert_w<<<(int)(((size_t)128 * FC1_KP + 255) / 256), 256, 0, stream>>>(fc1_w, wgtB);
    conv2_pool<<<dim3((C2_OUT_H * C2_OUT_H + 255) / 256, 15, BATCH), 256, 0, stream>>>(
        buf1, conv2_w, conv2_b, actA);
    fc1_wmma<<<128, 256, 0, stream>>>(actA, wgtB, bufP);
    fc1_reduce<<<(BATCH * FC1_N + 255) / 256, 256, 0, stream>>>(bufP, buf2);
    head<<<BATCH, 128, 0, stream>>>(buf2, fc1_b, fc2_w, fc2_b, fc3_w, fc3_b, qparams, out);
}
